// SSIMLoss_32126355374108
// MI455X (gfx1250) — matrix operand order repaired and checked by reading, not compiled, not run on hardware
//
#include <hip/hip_runtime.h>

typedef __attribute__((ext_vector_type(16))) _Float16 v16h;
typedef __attribute__((ext_vector_type(8)))  _Float16 v8h;
typedef __attribute__((ext_vector_type(2)))  _Float16 v2h;
typedef __attribute__((ext_vector_type(8)))  float    v8f;

#define IMG_H 512
#define IMG_W 512
#define NIMG  48                     // B*C = 16*3
#define TILES_PER_IMG 1024           // 32*32 tiles of 16x16
#define NTILES (NIMG * TILES_PER_IMG)          // 49152
#define WAVES_PER_BLOCK 4
#define THREADS_PER_BLOCK (WAVES_PER_BLOCK * 32)
#define NBLOCKS (NTILES / WAVES_PER_BLOCK)     // 12288
#define NPIX 12582912.0f                       // 16*3*512*512

__global__ __launch_bounds__(THREADS_PER_BLOCK)
void ssim_tile_kernel(const float* __restrict__ pred,
                      const float* __restrict__ target,
                      const float* __restrict__ window,
                      float* __restrict__ partial)
{
    // 5 quantity patches (p, t, p^2, t^2, p*t), f16, 32x32 row-major, per wave
    __shared__ __align__(16) _Float16 sQ[5][WAVES_PER_BLOCK][32 * 32];
    __shared__ __align__(32) _Float16 sTmp[WAVES_PER_BLOCK][16 * 32]; // [col n][row k]
    __shared__ float sRed[WAVES_PER_BLOCK];

    const int tid  = threadIdx.x;
    const int wave = tid >> 5;
    const int lane = tid & 31;
    const int lh   = lane >> 4;   // lane half (0: lanes 0-15, 1: lanes 16-31)
    const int n    = lane & 15;   // col index for B/C/D, row index for A

    // Recover 1D gaussian exactly from the 2D window diagonal: w2d[j][j] = g[j]^2
    float g[11];
    #pragma unroll
    for (int j = 0; j < 11; ++j) g[j] = sqrtf(window[j * 11 + j]);
    auto gband = [&](int d) -> float { return (d >= 0 && d <= 10) ? g[d] : 0.0f; };

    // Horizontal coefficients Gh (32x16, B-layout): element j -> K = (j<8 ? j : j+8) + lh*8
    v16h bGh;
    #pragma unroll
    for (int j = 0; j < 16; ++j) {
        int K = (j < 8 ? j : j + 8) + lh * 8;
        bGh[j] = (_Float16)gband(K - n);
    }
    // Vertical coefficients Gv (16x32, A-layout): row n, element j -> K below
    v16h aGv;
    #pragma unroll
    for (int j = 0; j < 16; ++j) {
        int K = (j < 8 ? j : j + 8) + lh * 8;
        aGv[j] = (_Float16)gband(K - n);
    }

    const int tile = blockIdx.x * WAVES_PER_BLOCK + wave;
    const int img  = tile / TILES_PER_IMG;
    const int rem  = tile - img * TILES_PER_IMG;
    const int ty   = rem >> 5;
    const int tx   = rem & 31;
    const int y0 = ty * 16, x0 = tx * 16;
    const size_t base = (size_t)img * (IMG_H * IMG_W);

    // ---- stage zero-padded 32x32 f16 patches of all 5 quantities ----
    // Each lane owns a pair of adjacent columns; half-wave per row parity.
    {
        const int cx  = (lane & 15) * 2;
        const int ry  = lane >> 4;
        const int gx0 = x0 - 5 + cx;
        const int gx1 = gx0 + 1;
        const bool x0in = (gx0 >= 0) && (gx0 < IMG_W);
        const bool x1in = (gx1 >= 0) && (gx1 < IMG_W);
        #pragma unroll 4
        for (int it = 0; it < 16; ++it) {
            const int i  = it * 2 + ry;
            const int gy = y0 - 5 + i;
            const bool yin = (gy >= 0) && (gy < IMG_H) && (i < 26);
            const size_t off = base + (size_t)gy * IMG_W + gx0;
            const float p0 = (yin && x0in) ? pred[off]       : 0.0f;
            const float p1 = (yin && x1in) ? pred[off + 1]   : 0.0f;
            const float t0 = (yin && x0in) ? target[off]     : 0.0f;
            const float t1 = (yin && x1in) ? target[off + 1] : 0.0f;
            const int sidx = i * 32 + cx;
            *(v2h*)&sQ[0][wave][sidx] = (v2h){(_Float16)p0,        (_Float16)p1};
            *(v2h*)&sQ[1][wave][sidx] = (v2h){(_Float16)t0,        (_Float16)t1};
            *(v2h*)&sQ[2][wave][sidx] = (v2h){(_Float16)(p0 * p0), (_Float16)(p1 * p1)};
            *(v2h*)&sQ[3][wave][sidx] = (v2h){(_Float16)(t0 * t0), (_Float16)(t1 * t1)};
            *(v2h*)&sQ[4][wave][sidx] = (v2h){(_Float16)(p0 * t0), (_Float16)(p1 * t1)};
        }
    }
    // sQ/sTmp are wave-private: LDS ops from one wave are serviced in order
    // (DScnt in-order), so only compiler ordering is needed, not a block
    // barrier. The workgroup fence pins LDS op order and keeps waves decoupled.
    __threadfence_block();

    const float C1 = 1e-4f;   // 0.01^2
    const float C2 = 9e-4f;   // 0.03^2

    v8f MU[5];
    #pragma unroll
    for (int q = 0; q < 5; ++q) {
        // ---- horizontal pass: two 16-row halves of the 26-row patch ----
        v8f dd[2];
        #pragma unroll
        for (int hb = 0; hb < 2; ++hb) {
            // A-layout 16x32: lane row n, K runs {0..7,16..23} + lh*8 are two
            // contiguous 16B runs of the f16 patch row -> pure ds_load_b128.
            const _Float16* rowp = &sQ[q][wave][(hb * 16 + n) * 32];
            const v8h lo = *(const v8h*)(rowp + lh * 8);
            const v8h hi = *(const v8h*)(rowp + 16 + lh * 8);
            const v16h a = __builtin_shufflevector(lo, hi,
                0, 1, 2, 3, 4, 5, 6, 7, 8, 9, 10, 11, 12, 13, 14, 15);
            v8f z = {};
            dd[hb] = __builtin_amdgcn_wmma_f32_16x16x32_f16(
                false, a, false, bGh, (short)0, z, false, false);
        }

        // ---- transpose bounce through LDS: D layout -> B layout (f16, [n][k]) ----
        #pragma unroll
        for (int r = 0; r < 8; ++r) {
            const int row0 = r + lh * 8;          // rows 0..15
            const int row1 = 16 + r + lh * 8;     // rows 16..31
            sTmp[wave][n * 32 + row0] = (_Float16)dd[0][r];
            sTmp[wave][n * 32 + row1] = (_Float16)dd[1][r];
        }
        __threadfence_block();   // order store->reload; in-wave DS is in-order

        const v8h* tp = (const v8h*)&sTmp[wave][n * 32 + lh * 8];
        const v8h blo = tp[0];
        const v8h bhi = tp[2];
        const v16h Bt = __builtin_shufflevector(blo, bhi,
            0, 1, 2, 3, 4, 5, 6, 7, 8, 9, 10, 11, 12, 13, 14, 15);

        // ---- vertical pass ----
        v8f z = {};
        MU[q] = __builtin_amdgcn_wmma_f32_16x16x32_f16(
            false, aGv, false, Bt, (short)0, z, false, false);
        // sTmp reuse next iteration is safe: these loads precede the next
        // stores in wave program order and DS ops stay in order per wave.
        __threadfence_block();
    }

    // ---- SSIM map + per-lane partial sum (8 pixels per lane, D layout) ----
    float acc = 0.0f;
    #pragma unroll
    for (int r = 0; r < 8; ++r) {
        const float mu1 = MU[0][r], mu2 = MU[1][r];
        const float bp2 = MU[2][r], bt2 = MU[3][r], bpt = MU[4][r];
        const float mu1s = mu1 * mu1;
        const float mu2s = mu2 * mu2;
        const float mu12 = mu1 * mu2;
        const float s1  = bp2 - mu1s;
        const float s2  = bt2 - mu2s;
        const float s12 = bpt - mu12;
        const float num = (2.0f * mu12 + C1) * (2.0f * s12 + C2);
        const float den = (mu1s + mu2s + C1) * (s1 + s2 + C2);
        acc += num / den;
    }

    // ---- deterministic reduction: wave32 butterfly, then cross-wave ----
    #pragma unroll
    for (int m = 16; m > 0; m >>= 1) acc += __shfl_xor(acc, m, 32);
    if (lane == 0) sRed[wave] = acc;
    __syncthreads();
    if (tid == 0) {
        float s = 0.0f;
        #pragma unroll
        for (int w = 0; w < WAVES_PER_BLOCK; ++w) s += sRed[w];
        partial[blockIdx.x] = s;
    }
}

__global__ __launch_bounds__(256)
void ssim_reduce_kernel(const float* __restrict__ partial, float* __restrict__ out)
{
    __shared__ float s[256];
    float a = 0.0f;
    for (int i = threadIdx.x; i < NBLOCKS; i += 256) a += partial[i];
    s[threadIdx.x] = a;
    __syncthreads();
    #pragma unroll
    for (int st = 128; st > 0; st >>= 1) {
        if (threadIdx.x < st) s[threadIdx.x] += s[threadIdx.x + st];
        __syncthreads();
    }
    if (threadIdx.x == 0) out[0] = 1.0f - s[0] / NPIX;
}

extern "C" void kernel_launch(void* const* d_in, const int* in_sizes, int n_in,
                              void* d_out, int out_size, void* d_ws, size_t ws_size,
                              hipStream_t stream) {
    const float* pred   = (const float*)d_in[0];
    const float* target = (const float*)d_in[1];
    const float* window = (const float*)d_in[2];
    float* partial = (float*)d_ws;   // NBLOCKS floats of scratch

    ssim_tile_kernel<<<NBLOCKS, THREADS_PER_BLOCK, 0, stream>>>(pred, target, window, partial);
    ssim_reduce_kernel<<<1, 256, 0, stream>>>(partial, (float*)d_out);
}
